// PET_Randoms_GNN_7035156431399
// MI455X (gfx1250) — matrix operand order repaired and checked by reading, not compiled, not run on hardware
//
#include <hip/hip_runtime.h>
#include <hip/hip_bf16.h>

#define NN 50000
#define EE 800000
#define DD 128

typedef __attribute__((ext_vector_type(16))) __bf16 v16bf;
typedef __attribute__((ext_vector_type(8)))  float  v8f;

union FragU { v16bf v; uint4 q[2]; };

// A fragment: 16x32 bf16, A row-major. Lane L<16: row M=L, K = k0..k0+15
// Lane L>=16: row M=L-16, K = k0+16..k0+31. Two b128 loads per lane.
__device__ __forceinline__ v16bf load_fragA_ptr(const __bf16* rowp, int k0, int lane) {
  int kk = k0 + ((lane >> 4) << 4);
  FragU f;
  f.q[0] = *(const uint4*)(rowp + kk);
  f.q[1] = *(const uint4*)(rowp + kk + 8);
  return f.v;
}

// B fragment from pre-transposed Bt [N,K] row-major. Lane L<16: col N=L, K=k0..k0+15;
// Lane L>=16: col N=L-16, K=k0+16..k0+31. Two b128 loads per lane.
__device__ __forceinline__ v16bf load_fragB(const __bf16* Bt, int ldb, int n0, int k0, int lane) {
  int n = n0 + (lane & 15);
  int kk = k0 + ((lane >> 4) << 4);
  const __bf16* p = Bt + (size_t)n * ldb + kk;
  FragU f;
  f.q[0] = *(const uint4*)(p);
  f.q[1] = *(const uint4*)(p + 8);
  return f.v;
}

#define WMMA_BF16(A_, B_, C_) \
  __builtin_amdgcn_wmma_f32_16x16x32_bf16(false, (A_), false, (B_), (short)0, (C_), false, false)

// monotonic float <-> uint for atomicMax-based segment max
__device__ __forceinline__ unsigned f2ord(float f) {
  unsigned u = __float_as_uint(f);
  return (u & 0x80000000u) ? ~u : (u | 0x80000000u);
}
__device__ __forceinline__ float ord2f(unsigned u) {
  unsigned b = (u & 0x80000000u) ? (u & 0x7fffffffu) : ~u;
  return __uint_as_float(b);
}

// ---------------- small prep kernels ----------------

__global__ void k_transpose_bf16(const float* __restrict__ B, __bf16* __restrict__ Bt,
                                 int K, int N, int ldb) {
  int idx = blockIdx.x * blockDim.x + threadIdx.x;
  if (idx >= K * N) return;
  int k = idx / N, n = idx % N;
  Bt[(size_t)n * ldb + k] = (__bf16)B[idx];
}

__global__ void k_encoder(const float* __restrict__ x, const float* __restrict__ W,
                          const float* __restrict__ b, __bf16* __restrict__ hb) {
  int idx = blockIdx.x * blockDim.x + threadIdx.x;
  if (idx >= NN * DD) return;
  int node = idx >> 7, d = idx & 127;
  const float* xr = x + (size_t)node * 5;
  float s = b[d];
#pragma unroll
  for (int f = 0; f < 5; ++f) s += xr[f] * W[f * DD + d];
  s = fmaxf(s, 0.f);
  hb[idx] = (__bf16)s;
}

__global__ void k_edge_feats(const float* __restrict__ x, const int* __restrict__ erow,
                             const int* __restrict__ ecol, float* __restrict__ dt,
                             float* __restrict__ dist) {
  int e = blockIdx.x * blockDim.x + threadIdx.x;
  if (e >= EE) return;
  int r = erow[e], c = ecol[e];
  const float* xr = x + (size_t)r * 5;
  const float* xc = x + (size_t)c * 5;
  dt[e] = fabsf(xr[4] - xc[4]);
  float dx = xr[1] - xc[1], dy = xr[2] - xc[2], dz = xr[3] - xc[3];
  dist[e] = sqrtf(dx * dx + dy * dy + dz * dz);
}

__global__ void k_fill_f32(float* p, float v, int n) {
  int i = blockIdx.x * blockDim.x + threadIdx.x;
  if (i < n) p[i] = v;
}

// ---------------- dense GEMM: C = A(bf16)[M,K] x Bt(bf16)[N,K]^T + bias, f32 out ----------------
// one wave = 16x64 output tile: 1 A fragment x 4 B fragments per k-step

__global__ void k_gemm_bf16_f32(const __bf16* __restrict__ A, const __bf16* __restrict__ Bt,
                                const float* __restrict__ bias, float* __restrict__ C,
                                int M, int N, int K) {
  int lane = threadIdx.x & 31;
  int wave = threadIdx.x >> 5;
  int tiles_n = N >> 6;
  int tile = blockIdx.x * 8 + wave;
  if (tile >= (M >> 4) * tiles_n) return;
  int m0 = (tile / tiles_n) << 4;
  int n0 = (tile % tiles_n) << 6;
  const __bf16* rowp = A + (size_t)(m0 + (lane & 15)) * K;
  v8f acc0 = {}, acc1 = {}, acc2 = {}, acc3 = {};
  for (int k0 = 0; k0 < K; k0 += 32) {
    __builtin_prefetch(rowp + k0 + 32, 0, 0);
    v16bf a  = load_fragA_ptr(rowp, k0, lane);
    v16bf b0 = load_fragB(Bt, K, n0, k0, lane);
    v16bf b1 = load_fragB(Bt, K, n0 + 16, k0, lane);
    v16bf b2 = load_fragB(Bt, K, n0 + 32, k0, lane);
    v16bf b3 = load_fragB(Bt, K, n0 + 48, k0, lane);
    acc0 = WMMA_BF16(a, b0, acc0);
    acc1 = WMMA_BF16(a, b1, acc1);
    acc2 = WMMA_BF16(a, b2, acc2);
    acc3 = WMMA_BF16(a, b3, acc3);
  }
  int n = lane & 15;
  int rb = (lane >> 4) << 3;
  v8f accs[4] = {acc0, acc1, acc2, acc3};
#pragma unroll
  for (int j = 0; j < 4; ++j) {
    int nc = n0 + j * 16 + n;
    float bv = bias[nc];
#pragma unroll
    for (int i = 0; i < 8; ++i)
      C[(size_t)(m0 + rb + i) * N + nc] = accs[j][i] + bv;
  }
}

// ---------------- GATv2 edge phase ----------------

__global__ void k_edge_alpha(const float* __restrict__ xl, const float* __restrict__ xr,
                             const int* __restrict__ erow, const int* __restrict__ ecol,
                             const float* __restrict__ att, float* __restrict__ alpha,
                             unsigned* __restrict__ amax) {
  int idx = blockIdx.x * blockDim.x + threadIdx.x;
  int e = idx >> 2, h = idx & 3;
  if (e >= EE + NN) return;
  int s = (e < EE) ? erow[e] : (e - EE);
  int d = (e < EE) ? ecol[e] : (e - EE);
  const float* pl = xl + (size_t)s * DD + h * 32;
  const float* pr = xr + (size_t)d * DD + h * 32;
  const float* pa = att + h * 32;
  float a = 0.f;
#pragma unroll
  for (int c = 0; c < 32; ++c) {
    float v = pl[c] + pr[c];
    v = (v > 0.f) ? v : 0.2f * v;
    a += v * pa[c];
  }
  alpha[(size_t)e * 4 + h] = a;
  atomicMax(&amax[d * 4 + h], f2ord(a));
}

__global__ void k_edge_expsum(float* __restrict__ alpha, const unsigned* __restrict__ amax,
                              const int* __restrict__ ecol, float* __restrict__ denom) {
  int idx = blockIdx.x * blockDim.x + threadIdx.x;
  int e = idx >> 2, h = idx & 3;
  if (e >= EE + NN) return;
  int d = (e < EE) ? ecol[e] : (e - EE);
  float m = ord2f(amax[d * 4 + h]);
  float v = __expf(alpha[(size_t)e * 4 + h] - m);
  alpha[(size_t)e * 4 + h] = v;
  atomicAdd(&denom[d * 4 + h], v);
}

__global__ void k_edge_scatter(const float* __restrict__ alpha, const float* __restrict__ denom,
                               const float* __restrict__ xl, const int* __restrict__ erow,
                               const int* __restrict__ ecol, float* __restrict__ agg) {
  int idx = blockIdx.x * blockDim.x + threadIdx.x;
  int e = idx >> 2, h = idx & 3;
  if (e >= EE + NN) return;
  int s = (e < EE) ? erow[e] : (e - EE);
  int d = (e < EE) ? ecol[e] : (e - EE);
  float w = alpha[(size_t)e * 4 + h] / (denom[d * 4 + h] + 1e-16f);
  const float* pl = xl + (size_t)s * DD + h * 32;
  float* po = agg + (size_t)d * DD + h * 32;
#pragma unroll
  for (int c = 0; c < 32; ++c) atomicAdd(&po[c], pl[c] * w);
}

__global__ void k_node_final(const float* __restrict__ agg, const float* __restrict__ gbias,
                             __bf16* __restrict__ hb) {
  int idx = blockIdx.x * blockDim.x + threadIdx.x;
  if (idx >= NN * DD) return;
  float v = agg[idx] + gbias[idx & 127];
  v = (v > 0.f) ? v : (__expf(v) - 1.f);   // ELU
  hb[idx] = (__bf16)v;
}

// ---------------- classifier layer 1: gathered [E,256+2] x c1, 16x64 tiles ----------------

__global__ void k_gemm_c1(const __bf16* __restrict__ hb, const int* __restrict__ erow,
                          const int* __restrict__ ecol, const __bf16* __restrict__ c1t,
                          const float* __restrict__ b1, const float* __restrict__ dt,
                          const float* __restrict__ dist, __bf16* __restrict__ o1b) {
  const int ldb = 264;  // 258 padded to 16B-aligned rows
  int lane = threadIdx.x & 31;
  int wave = threadIdx.x >> 5;
  int tile = blockIdx.x * 8 + wave;
  if (tile >= (EE >> 4) * 2) return;
  int m0 = (tile >> 1) << 4;
  int n0 = (tile & 1) << 6;
  int e = m0 + (lane & 15);
  const __bf16* rowS = hb + (size_t)erow[e] * DD;
  const __bf16* rowC = hb + (size_t)ecol[e] * DD;
  v8f acc0 = {}, acc1 = {}, acc2 = {}, acc3 = {};
  for (int k0 = 0; k0 < 128; k0 += 32) {
    v16bf a  = load_fragA_ptr(rowS, k0, lane);
    v16bf b0 = load_fragB(c1t, ldb, n0, k0, lane);
    v16bf b1 = load_fragB(c1t, ldb, n0 + 16, k0, lane);
    v16bf b2 = load_fragB(c1t, ldb, n0 + 32, k0, lane);
    v16bf b3 = load_fragB(c1t, ldb, n0 + 48, k0, lane);
    acc0 = WMMA_BF16(a, b0, acc0);
    acc1 = WMMA_BF16(a, b1, acc1);
    acc2 = WMMA_BF16(a, b2, acc2);
    acc3 = WMMA_BF16(a, b3, acc3);
  }
  for (int k0 = 0; k0 < 128; k0 += 32) {
    v16bf a  = load_fragA_ptr(rowC, k0, lane);
    v16bf b0 = load_fragB(c1t, ldb, n0, 128 + k0, lane);
    v16bf b1 = load_fragB(c1t, ldb, n0 + 16, 128 + k0, lane);
    v16bf b2 = load_fragB(c1t, ldb, n0 + 32, 128 + k0, lane);
    v16bf b3 = load_fragB(c1t, ldb, n0 + 48, 128 + k0, lane);
    acc0 = WMMA_BF16(a, b0, acc0);
    acc1 = WMMA_BF16(a, b1, acc1);
    acc2 = WMMA_BF16(a, b2, acc2);
    acc3 = WMMA_BF16(a, b3, acc3);
  }
  int n = lane & 15;
  int rb = (lane >> 4) << 3;
  v8f accs[4] = {acc0, acc1, acc2, acc3};
#pragma unroll
  for (int j = 0; j < 4; ++j) {
    int nc = n0 + j * 16 + n;
    float bv = b1[nc];
    float wdt = (float)c1t[(size_t)nc * ldb + 256];
    float wds = (float)c1t[(size_t)nc * ldb + 257];
#pragma unroll
    for (int i = 0; i < 8; ++i) {
      int ei = m0 + rb + i;
      float v = accs[j][i] + bv + dt[ei] * wdt + dist[ei] * wds;
      v = fmaxf(v, 0.f);
      o1b[(size_t)ei * DD + nc] = (__bf16)v;
    }
  }
}

// ------- classifier layers 2+3 fused: out[e] = relu(o1[e] x c2 + b2) . c3 + c3b --------
// one wave owns the full 64-wide layer-2 row -> in-wave reduction, plain store

__global__ void k_gemm_c2_c3(const __bf16* __restrict__ o1b, const __bf16* __restrict__ c2t,
                             const float* __restrict__ b2, const float* __restrict__ c3w,
                             const float* __restrict__ c3b, float* __restrict__ out) {
  int lane = threadIdx.x & 31;
  int wave = threadIdx.x >> 5;
  int tile = blockIdx.x * 8 + wave;
  if (tile >= (EE >> 4)) return;
  int m0 = tile << 4;
  const __bf16* rowp = o1b + (size_t)(m0 + (lane & 15)) * DD;
  v8f acc0 = {}, acc1 = {}, acc2 = {}, acc3 = {};
  for (int k0 = 0; k0 < 128; k0 += 32) {
    v16bf a  = load_fragA_ptr(rowp, k0, lane);
    v16bf b0 = load_fragB(c2t, DD, 0, k0, lane);
    v16bf b1 = load_fragB(c2t, DD, 16, k0, lane);
    v16bf b2 = load_fragB(c2t, DD, 32, k0, lane);
    v16bf b3 = load_fragB(c2t, DD, 48, k0, lane);
    acc0 = WMMA_BF16(a, b0, acc0);
    acc1 = WMMA_BF16(a, b1, acc1);
    acc2 = WMMA_BF16(a, b2, acc2);
    acc3 = WMMA_BF16(a, b3, acc3);
  }
  int n = lane & 15;
  int rb = (lane >> 4) << 3;
  float bb[4] = {b2[n], b2[n + 16], b2[n + 32], b2[n + 48]};
  float cc[4] = {c3w[n], c3w[n + 16], c3w[n + 32], c3w[n + 48]};
  v8f accs[4] = {acc0, acc1, acc2, acc3};
  float part[8];
#pragma unroll
  for (int i = 0; i < 8; ++i) {
    float s = 0.f;
#pragma unroll
    for (int j = 0; j < 4; ++j) s += fmaxf(accs[j][i] + bb[j], 0.f) * cc[j];
    part[i] = s;
  }
  // reduce the 16 N-lanes of each half-wave (the two halves hold different rows)
#pragma unroll
  for (int m = 1; m < 16; m <<= 1) {
#pragma unroll
    for (int i = 0; i < 8; ++i) part[i] += __shfl_xor(part[i], m, 32);
  }
  if (n == 0) {
    float b3v = c3b[0];
#pragma unroll
    for (int i = 0; i < 8; ++i) out[m0 + rb + i] = part[i] + b3v;
  }
}

// ---------------- host orchestration ----------------

extern "C" void kernel_launch(void* const* d_in, const int* in_sizes, int n_in,
                              void* d_out, int out_size, void* d_ws, size_t ws_size,
                              hipStream_t stream) {
  const float* x     = (const float*)d_in[0];
  const int*   eidx  = (const int*)d_in[1];
  const float* encW  = (const float*)d_in[2];
  const float* encB  = (const float*)d_in[3];
  const float* Wl    = (const float*)d_in[4];
  const float* bl    = (const float*)d_in[5];
  const float* Wr    = (const float*)d_in[6];
  const float* br    = (const float*)d_in[7];
  const float* att   = (const float*)d_in[8];
  const float* gbias = (const float*)d_in[9];
  const float* c1W   = (const float*)d_in[10];
  const float* c1b   = (const float*)d_in[11];
  const float* c2W   = (const float*)d_in[12];
  const float* c2b   = (const float*)d_in[13];
  const float* c3W   = (const float*)d_in[14];
  const float* c3b   = (const float*)d_in[15];
  const int* erow = eidx;
  const int* ecol = eidx + EE;

  char* w = (char*)d_ws;
  auto take = [&](size_t bytes) -> char* {
    char* p = w;
    w += (bytes + 255) & ~(size_t)255;
    return p;
  };
  __bf16*   hb    = (__bf16*)  take((size_t)NN * DD * 2);
  float*    xl    = (float*)   take((size_t)NN * DD * 4);
  float*    xr    = (float*)   take((size_t)NN * DD * 4);
  float*    agg   = (float*)   take((size_t)NN * DD * 4);
  float*    alpha = (float*)   take((size_t)(EE + NN) * 4 * 4);
  unsigned* amax  = (unsigned*)take((size_t)NN * 4 * 4);
  float*    denom = (float*)   take((size_t)NN * 4 * 4);
  float*    dt    = (float*)   take((size_t)EE * 4);
  float*    dist  = (float*)   take((size_t)EE * 4);
  __bf16*   WlT   = (__bf16*)  take((size_t)3 * DD * DD * 2);
  __bf16*   WrT   = (__bf16*)  take((size_t)3 * DD * DD * 2);
  __bf16*   c1T   = (__bf16*)  take((size_t)DD * 264 * 2);
  __bf16*   c2T   = (__bf16*)  take((size_t)64 * DD * 2);
  __bf16*   o1b   = (__bf16*)  take((size_t)EE * DD * 2);

  // weight prep (bf16 transposed)
  for (int l = 0; l < 3; ++l) {
    k_transpose_bf16<<<(DD * DD + 255) / 256, 256, 0, stream>>>(
        Wl + (size_t)l * DD * DD, WlT + (size_t)l * DD * DD, DD, DD, DD);
    k_transpose_bf16<<<(DD * DD + 255) / 256, 256, 0, stream>>>(
        Wr + (size_t)l * DD * DD, WrT + (size_t)l * DD * DD, DD, DD, DD);
  }
  k_transpose_bf16<<<(258 * DD + 255) / 256, 256, 0, stream>>>(c1W, c1T, 258, DD, 264);
  k_transpose_bf16<<<(DD * 64 + 255) / 256, 256, 0, stream>>>(c2W, c2T, DD, 64, DD);

  // encoder + edge geometric features
  k_encoder<<<(NN * DD + 255) / 256, 256, 0, stream>>>(x, encW, encB, hb);
  k_edge_feats<<<(EE + 255) / 256, 256, 0, stream>>>(x, erow, ecol, dt, dist);

  const int ETH = (EE + NN) * 4;
  const int gemm_node_blocks = ((NN / 16) * (DD / 64) + 7) / 8;   // 6250 tiles -> 782 blocks
  for (int l = 0; l < 3; ++l) {
    k_gemm_bf16_f32<<<gemm_node_blocks, 256, 0, stream>>>(
        hb, WlT + (size_t)l * DD * DD, bl + l * DD, xl, NN, DD, DD);
    k_gemm_bf16_f32<<<gemm_node_blocks, 256, 0, stream>>>(
        hb, WrT + (size_t)l * DD * DD, br + l * DD, xr, NN, DD, DD);
    k_fill_f32<<<(NN * 4 + 255) / 256, 256, 0, stream>>>((float*)amax, 0.f, NN * 4);
    k_fill_f32<<<(NN * 4 + 255) / 256, 256, 0, stream>>>(denom, 0.f, NN * 4);
    k_fill_f32<<<(NN * DD + 255) / 256, 256, 0, stream>>>(agg, 0.f, NN * DD);
    k_edge_alpha<<<(ETH + 255) / 256, 256, 0, stream>>>(xl, xr, erow, ecol,
                                                        att + l * 4 * 32, alpha, amax);
    k_edge_expsum<<<(ETH + 255) / 256, 256, 0, stream>>>(alpha, amax, ecol, denom);
    k_edge_scatter<<<(ETH + 255) / 256, 256, 0, stream>>>(alpha, denom, xl, erow, ecol, agg);
    k_node_final<<<(NN * DD + 255) / 256, 256, 0, stream>>>(agg, gbias + l * DD, hb);
  }

  // edge classifier
  k_gemm_c1<<<((EE / 16) * 2 + 7) / 8, 256, 0, stream>>>(hb, erow, ecol, c1T, c1b, dt,
                                                         dist, o1b);
  k_gemm_c2_c3<<<((EE / 16) + 7) / 8, 256, 0, stream>>>(o1b, c2T, c2b, c3W, c3b,
                                                        (float*)d_out);
}
